// Decoder_66099546685445
// MI455X (gfx1250) — compile-verified, matrix operand order checked by reading, not run on hardware
//
#include <hip/hip_runtime.h>
#include <math.h>

// ---------------------------------------------------------------------------
// Pointer-network decoder for MI455X (gfx1250, wave32, WMMA).
//   B=1024, S=100, E=H=128.
// Strategy:
//  * One-time precompute of e_g/e_p (105 MB total -> resident in 192 MB L2).
//  * One-time f32->f16 weight swizzle into the exact WMMA B-operand lane
//    layout (Wi/Wh/g_Wq/p_Wq, ~320 KB f16, L2-resident) so GEMM inner loops
//    issue pure b128 loads + v_wmma with no per-step cvt of weights.
//  * Per step: WMMA f16 GEMMs (LSTM gates, both attention query projections),
//    then VALU tanh/softmax kernels reading e_g/e_p out of L2.
// ---------------------------------------------------------------------------

#define Bsz 1024
#define Ssz 100
#define Hsz 128
#define Esz 128

typedef __attribute__((ext_vector_type(16))) _Float16 v16h;
typedef __attribute__((ext_vector_type(8)))  float    v8f;

__device__ __forceinline__ float fast_rcp(float x) {
  return __builtin_amdgcn_rcpf(x);
}
__device__ __forceinline__ float fast_sigmoid(float x) {
  return fast_rcp(1.0f + __expf(-x));
}
__device__ __forceinline__ float fast_tanh(float x) {
  x = fminf(fmaxf(x, -15.0f), 15.0f);
  float e2 = __expf(2.0f * x);
  return (e2 - 1.0f) * fast_rcp(e2 + 1.0f);
}

// Load a 16x32 f16 tile (A operand) for v_wmma_f32_16x16x32_f16 from a
// row-major f32 matrix, converting in-register. Per ISA 16-bit A layout:
// lane l holds row (l&15); v16h elems 0-7 = K kh..kh+7, elems 8-15 =
// K 16+kh..16+kh+7, kh = (l>>4)*8.
__device__ __forceinline__ v16h load_tile16(const float* __restrict__ base,
                                            int rowBase, int ld, int kb) {
  const int l  = threadIdx.x & 31;
  const int r  = rowBase + (l & 15);
  const int kh = (l >> 4) * 8;
  const float* p = base + (size_t)r * ld + kb + kh;
  float4 a0 = *reinterpret_cast<const float4*>(p);
  float4 a1 = *reinterpret_cast<const float4*>(p + 4);
  float4 b0 = *reinterpret_cast<const float4*>(p + 16);
  float4 b1 = *reinterpret_cast<const float4*>(p + 20);
  v16h o;
  o[0]  = (_Float16)a0.x; o[1]  = (_Float16)a0.y;
  o[2]  = (_Float16)a0.z; o[3]  = (_Float16)a0.w;
  o[4]  = (_Float16)a1.x; o[5]  = (_Float16)a1.y;
  o[6]  = (_Float16)a1.z; o[7]  = (_Float16)a1.w;
  o[8]  = (_Float16)b0.x; o[9]  = (_Float16)b0.y;
  o[10] = (_Float16)b0.z; o[11] = (_Float16)b0.w;
  o[12] = (_Float16)b1.x; o[13] = (_Float16)b1.y;
  o[14] = (_Float16)b1.z; o[15] = (_Float16)b1.w;
  return o;
}

// Load a pre-swizzled f16 weight tile: tiles stored tile-major
// (tile = (n>>4)*4 + (kb>>5), K fixed at 128), 32 lanes x 16 halves each.
// One 32-byte aligned v16h load per lane -> two global_load_b128, no cvt.
__device__ __forceinline__ v16h load_wtile16(const unsigned short* __restrict__ wsw,
                                             int nb, int kb) {
  const int l = threadIdx.x & 31;
  const size_t tile = (size_t)(nb >> 4) * 4 + (kb >> 5);
  return *reinterpret_cast<const v16h*>(
      reinterpret_cast<const _Float16*>(wsw) + tile * 512 + (size_t)l * 16);
}

#define WMMA_F16(a, b, c) \
  __builtin_amdgcn_wmma_f32_16x16x32_f16(false, (a), false, (b), (short)0, (c), false, false)

// ---------------------------------------------------------------------------
// Init: zero the visited-mask (must run every kernel_launch call).
// ---------------------------------------------------------------------------
__global__ void init_mask_kernel(int* __restrict__ mask) {
  int i = blockIdx.x * 256 + threadIdx.x;
  if (i < Bsz * Ssz) mask[i] = 0;
}

// ---------------------------------------------------------------------------
// One-time: convert a row-major f32 weight matrix (nrows x 128) into f16
// tiles in the WMMA B-operand per-lane layout. One wave per 16x32 tile.
// grid covers ntiles_total = (nrows/16)*4 tiles, 8 waves per block.
// ---------------------------------------------------------------------------
__global__ __launch_bounds__(256) void swizzle_w_kernel(
    const float* __restrict__ W, unsigned short* __restrict__ out,
    int ntiles_total) {
  const int wid  = threadIdx.x >> 5;
  const int tile = blockIdx.x * 8 + wid;
  if (tile >= ntiles_total) return;
  const int nt = tile >> 2;        // n-tile index
  const int kc = tile & 3;         // k-chunk index
  const int l  = threadIdx.x & 31;
  const int r  = nt * 16 + (l & 15);
  const int kb = kc * 32 + (l >> 4) * 8;
  const float* p = W + (size_t)r * 128 + kb;
  _Float16* o = reinterpret_cast<_Float16*>(out) + (size_t)tile * 512 + (size_t)l * 16;
  #pragma unroll
  for (int j = 0; j < 8; ++j) {
    o[j]     = (_Float16)p[j];
    o[8 + j] = (_Float16)p[16 + j];
  }
}

// ---------------------------------------------------------------------------
// Precompute e[b][s][o] = sum_h W[o,h]*ctx[s,h,b] + bias[o] for both heads.
// ctx: (S,H,B). Output layout (B,S,H) so attention reads are contiguous in h.
// grid = (S, B/32, 2), block = 256.
// ---------------------------------------------------------------------------
__global__ __launch_bounds__(256) void precompute_e_kernel(
    const float* __restrict__ ctx,
    const float* __restrict__ Wg, const float* __restrict__ bg,
    const float* __restrict__ Wp, const float* __restrict__ bp,
    float* __restrict__ eg, float* __restrict__ ep) {
  const int s    = blockIdx.x;
  const int b0   = blockIdx.y * 32;
  const int head = blockIdx.z;
  const float* W    = head ? Wp : Wg;
  const float* bias = head ? bp : bg;
  float*       e    = head ? ep : eg;

  __shared__ float lds[Hsz][32];  // ctx[s, :, b0..b0+31]
  #pragma unroll
  for (int it = 0; it < 16; ++it) {
    int elem = it * 256 + threadIdx.x;
    int h = elem >> 5, bi = elem & 31;
    lds[h][bi] = ctx[((size_t)s * Hsz + h) * Bsz + b0 + bi];
  }
  __syncthreads();

  const int o  = threadIdx.x & 127;
  const int bg2 = (threadIdx.x >> 7) * 16;  // which half of the 32 b's
  float acc[16];
  #pragma unroll
  for (int j = 0; j < 16; ++j) acc[j] = 0.0f;
  for (int h = 0; h < Hsz; ++h) {
    float w = W[o * Hsz + h];
    #pragma unroll
    for (int j = 0; j < 16; ++j) acc[j] += w * lds[h][bg2 + j];
  }
  float bb = bias[o];
  #pragma unroll
  for (int j = 0; j < 16; ++j) {
    e[((size_t)(b0 + bg2 + j) * Ssz + s) * Hsz + o] = acc[j] + bb;
  }
}

// ---------------------------------------------------------------------------
// LSTM step: gates = x@Wi^T + bi + h@Wh^T + bh ; cy, hy.
// One wave computes a 16(b) x 16(h) tile of hy/cy via 4 WMMA accumulators
// (i,f,g,o gate columns share the same A tiles). Weights pre-swizzled f16.
// grid = 64 blocks x 8 waves = 512 tiles (64 M-tiles x 8 N-tiles).
// ---------------------------------------------------------------------------
__global__ __launch_bounds__(256) void lstm_kernel(
    const float* __restrict__ x, const float* __restrict__ hprev,
    const float* __restrict__ cprev,
    const unsigned short* __restrict__ Wi16, const float* __restrict__ bi,
    const unsigned short* __restrict__ Wh16, const float* __restrict__ bh,
    float* __restrict__ hout, float* __restrict__ cout) {
  const int wid = threadIdx.x >> 5;
  const int id  = blockIdx.x * 8 + wid;
  const int mb  = (id >> 3) * 16;  // batch-row base
  const int nb  = (id & 7) * 16;   // hidden-col base

  v8f ai = {0,0,0,0,0,0,0,0};
  v8f af = {0,0,0,0,0,0,0,0};
  v8f ag = {0,0,0,0,0,0,0,0};
  v8f ao = {0,0,0,0,0,0,0,0};

  #pragma unroll
  for (int kb = 0; kb < Esz; kb += 32) {  // x @ Wi^T
    v16h a  = load_tile16(x, mb, Esz, kb);
    v16h b0 = load_wtile16(Wi16, nb,       kb);
    v16h b1 = load_wtile16(Wi16, nb + 128, kb);
    v16h b2 = load_wtile16(Wi16, nb + 256, kb);
    v16h b3 = load_wtile16(Wi16, nb + 384, kb);
    ai = WMMA_F16(a, b0, ai);
    af = WMMA_F16(a, b1, af);
    ag = WMMA_F16(a, b2, ag);
    ao = WMMA_F16(a, b3, ao);
  }
  #pragma unroll
  for (int kb = 0; kb < Hsz; kb += 32) {  // h @ Wh^T
    v16h a  = load_tile16(hprev, mb, Hsz, kb);
    v16h b0 = load_wtile16(Wh16, nb,       kb);
    v16h b1 = load_wtile16(Wh16, nb + 128, kb);
    v16h b2 = load_wtile16(Wh16, nb + 256, kb);
    v16h b3 = load_wtile16(Wh16, nb + 384, kb);
    ai = WMMA_F16(a, b0, ai);
    af = WMMA_F16(a, b1, af);
    ag = WMMA_F16(a, b2, ag);
    ao = WMMA_F16(a, b3, ao);
  }

  // C layout: VGPR e, lane l -> row = mb + e + 8*(l>>4), col = nb + (l&15).
  const int l    = threadIdx.x & 31;
  const int n    = nb + (l & 15);
  const int mofs = (l >> 4) * 8;
  const float bii = bi[n]       + bh[n];
  const float bif = bi[n + 128] + bh[n + 128];
  const float big = bi[n + 256] + bh[n + 256];
  const float bio = bi[n + 384] + bh[n + 384];
  #pragma unroll
  for (int e = 0; e < 8; ++e) {
    const int row = mb + mofs + e;
    float gi = fast_sigmoid(ai[e] + bii);
    float gf = fast_sigmoid(af[e] + bif);
    float gg = fast_tanh   (ag[e] + big);
    float go = fast_sigmoid(ao[e] + bio);
    float cp = cprev[(size_t)row * Hsz + n];
    float cy = gf * cp + gi * gg;
    float hy = go * fast_tanh(cy);
    cout[(size_t)row * Hsz + n] = cy;
    hout[(size_t)row * Hsz + n] = hy;
  }
}

// ---------------------------------------------------------------------------
// Generic 1024x128x128 GEMM + bias: out = A @ W^T + bias (W pre-swizzled f16).
// grid = 64 blocks x 8 waves = 512 tiles.
// ---------------------------------------------------------------------------
__global__ __launch_bounds__(256) void gemm128_bias_kernel(
    const float* __restrict__ A, const unsigned short* __restrict__ W16,
    const float* __restrict__ bias, float* __restrict__ out) {
  const int wid = threadIdx.x >> 5;
  const int id  = blockIdx.x * 8 + wid;
  const int mb  = (id >> 3) * 16;
  const int nb  = (id & 7) * 16;
  v8f acc = {0,0,0,0,0,0,0,0};
  #pragma unroll
  for (int kb = 0; kb < Hsz; kb += 32) {
    v16h a = load_tile16(A, mb, Hsz, kb);
    v16h b = load_wtile16(W16, nb, kb);
    acc = WMMA_F16(a, b, acc);
  }
  const int l    = threadIdx.x & 31;
  const int n    = nb + (l & 15);
  const int mofs = (l >> 4) * 8;
  const float bb = bias[n];
  #pragma unroll
  for (int e = 0; e < 8; ++e) {
    out[(size_t)(mb + mofs + e) * Hsz + n] = acc[e] + bb;
  }
}

// ---------------------------------------------------------------------------
// Glimpse: u[b,s] = sum_h v[h]*tanh(q[b,h]+eg[b,s,h]); softmax over s (masked);
// gl[b,h] = sum_s p[s]*eg[b,s,h].  One 128-thread block per batch element.
// ---------------------------------------------------------------------------
__global__ __launch_bounds__(128) void glimpse_kernel(
    const float* __restrict__ q, const float* __restrict__ eg,
    const float* __restrict__ v, const int* __restrict__ mask,
    float* __restrict__ gl) {
  const int b = blockIdx.x;
  const int t = threadIdx.x;
  __shared__ float q_s[Hsz], v_s[Hsz], u_s[Hsz], red[Hsz];

  q_s[t] = q[(size_t)b * Hsz + t];
  v_s[t] = v[t];
  __syncthreads();

  const int wid = t >> 5, l = t & 31;
  const float* erow = eg + (size_t)b * Ssz * Hsz;
  const float4 vv = *reinterpret_cast<const float4*>(&v_s[l * 4]);
  const float4 qq = *reinterpret_cast<const float4*>(&q_s[l * 4]);
  for (int s = wid; s < Ssz; s += 4) {
    float4 ee = *reinterpret_cast<const float4*>(erow + (size_t)s * Hsz + l * 4);
    float val = vv.x * fast_tanh(qq.x + ee.x) + vv.y * fast_tanh(qq.y + ee.y)
              + vv.z * fast_tanh(qq.z + ee.z) + vv.w * fast_tanh(qq.w + ee.w);
    #pragma unroll
    for (int off = 16; off; off >>= 1) val += __shfl_xor(val, off, 32);
    if (l == 0) u_s[s] = val;
  }
  __syncthreads();

  // masked softmax over s
  const float NEG = -__builtin_inff();
  bool valid = (t < Ssz) && (mask[b * Ssz + t] == 0);
  float x = valid ? u_s[t] : NEG;
  red[t] = x;
  __syncthreads();
  for (int o = 64; o; o >>= 1) {
    if (t < o) red[t] = fmaxf(red[t], red[t + o]);
    __syncthreads();
  }
  float mx = red[0];
  __syncthreads();
  float pe = valid ? __expf(x - mx) : 0.0f;
  red[t] = pe;
  __syncthreads();
  for (int o = 64; o; o >>= 1) {
    if (t < o) red[t] += red[t + o];
    __syncthreads();
  }
  float inv = fast_rcp(red[0]);
  __syncthreads();
  u_s[t] = pe * inv;
  __syncthreads();

  // gl[b,h] — coalesced: thread t = h, contiguous across the block per s.
  float acc = 0.0f;
  for (int s = 0; s < Ssz; ++s) acc += u_s[s] * erow[(size_t)s * Hsz + t];
  gl[(size_t)b * Hsz + t] = acc;
}

// ---------------------------------------------------------------------------
// Pointer head: u, logits = mask ? -inf : 10*tanh(u); softmax -> probs out;
// argmax (first max) -> idx; update mask; gather next x from embedded_inputs;
// write sels (as float).  One 128-thread block per batch element.
// ---------------------------------------------------------------------------
__global__ __launch_bounds__(128) void pointer_kernel(
    const float* __restrict__ q, const float* __restrict__ ep,
    const float* __restrict__ v, int* __restrict__ mask,
    const float* __restrict__ embedded, float* __restrict__ xnext,
    float* __restrict__ probs_out, float* __restrict__ sels_out, int tstep) {
  const int b = blockIdx.x;
  const int t = threadIdx.x;
  __shared__ float q_s[Hsz], v_s[Hsz], u_s[Hsz], red[Hsz];
  __shared__ int   redi[Hsz];

  q_s[t] = q[(size_t)b * Hsz + t];
  v_s[t] = v[t];
  __syncthreads();

  const int wid = t >> 5, l = t & 31;
  const float* erow = ep + (size_t)b * Ssz * Hsz;
  const float4 vv = *reinterpret_cast<const float4*>(&v_s[l * 4]);
  const float4 qq = *reinterpret_cast<const float4*>(&q_s[l * 4]);
  for (int s = wid; s < Ssz; s += 4) {
    float4 ee = *reinterpret_cast<const float4*>(erow + (size_t)s * Hsz + l * 4);
    float val = vv.x * fast_tanh(qq.x + ee.x) + vv.y * fast_tanh(qq.y + ee.y)
              + vv.z * fast_tanh(qq.z + ee.z) + vv.w * fast_tanh(qq.w + ee.w);
    #pragma unroll
    for (int off = 16; off; off >>= 1) val += __shfl_xor(val, off, 32);
    if (l == 0) u_s[s] = val;
  }
  __syncthreads();

  const float NEG = -__builtin_inff();
  bool inb   = (t < Ssz);
  bool valid = inb && (mask[b * Ssz + t] == 0);
  float lg = valid ? 10.0f * fast_tanh(u_s[t]) : NEG;

  red[t] = lg;
  __syncthreads();
  for (int o = 64; o; o >>= 1) {
    if (t < o) red[t] = fmaxf(red[t], red[t + o]);
    __syncthreads();
  }
  float mx = red[0];
  __syncthreads();
  float pe = valid ? __expf(lg - mx) : 0.0f;
  red[t] = pe;
  __syncthreads();
  for (int o = 64; o; o >>= 1) {
    if (t < o) red[t] += red[t + o];
    __syncthreads();
  }
  float inv = fast_rcp(red[0]);
  __syncthreads();
  float p = pe * inv;
  if (inb) probs_out[(size_t)tstep * Bsz * Ssz + (size_t)b * Ssz + t] = p;

  // argmax (first occurrence of max, matching jnp.argmax)
  red[t]  = inb ? p : -1.0f;
  redi[t] = t;
  __syncthreads();
  for (int o = 64; o; o >>= 1) {
    if (t < o) {
      float v2 = red[t + o];
      int   i2 = redi[t + o];
      if (v2 > red[t] || (v2 == red[t] && i2 < redi[t])) {
        red[t] = v2; redi[t] = i2;
      }
    }
    __syncthreads();
  }
  int idx = redi[0];

  if (t == 0) {
    mask[b * Ssz + idx] = 1;  // visited for next step
    sels_out[(size_t)tstep * Bsz + b] = (float)idx;
  }
  // next decoder input: embedded_inputs[idx, b, :]
  xnext[(size_t)b * Esz + t] =
      embedded[((size_t)idx * Bsz + b) * Esz + t];
}

// ---------------------------------------------------------------------------
// Host side
// ---------------------------------------------------------------------------
extern "C" void kernel_launch(void* const* d_in, const int* in_sizes, int n_in,
                              void* d_out, int out_size, void* d_ws, size_t ws_size,
                              hipStream_t stream) {
  const float* decoder_input = (const float*)d_in[0];
  const float* embedded      = (const float*)d_in[1];
  const float* h0            = (const float*)d_in[2];
  const float* c0            = (const float*)d_in[3];
  const float* context       = (const float*)d_in[4];
  const float* Wi            = (const float*)d_in[5];
  const float* bi            = (const float*)d_in[6];
  const float* Wh            = (const float*)d_in[7];
  const float* bh            = (const float*)d_in[8];
  const float* g_Wq          = (const float*)d_in[9];
  const float* g_bq          = (const float*)d_in[10];
  const float* g_Wref        = (const float*)d_in[11];
  const float* g_bref        = (const float*)d_in[12];
  const float* g_v           = (const float*)d_in[13];
  const float* p_Wq          = (const float*)d_in[14];
  const float* p_bq          = (const float*)d_in[15];
  const float* p_Wref        = (const float*)d_in[16];
  const float* p_bref        = (const float*)d_in[17];
  const float* p_v           = (const float*)d_in[18];

  float* out = (float*)d_out;

  // Workspace layout (floats then ints then f16). Total ~109.3 MB.
  float* ws = (float*)d_ws;
  const size_t N_E  = (size_t)Bsz * Ssz * Hsz;  // 13,107,200
  const size_t N_BH = (size_t)Bsz * Hsz;        // 131,072
  float* eg   = ws;                 // e_g  (B,S,H)
  float* ep   = eg  + N_E;          // e_p  (B,S,H)
  float* qbuf = ep  + N_E;          // query buffer (B,H)
  float* gl   = qbuf + N_BH;        // glimpse output (B,H)
  float* hb0  = gl  + N_BH;         // h double buffers
  float* hb1  = hb0 + N_BH;
  float* cb0  = hb1 + N_BH;         // c double buffers
  float* cb1  = cb0 + N_BH;
  float* xbuf = cb1 + N_BH;         // next decoder input (B,E)
  int*   mask = (int*)(xbuf + N_BH);          // (B,S) visited mask
  unsigned short* wi16 = (unsigned short*)(mask + Bsz * Ssz);  // 512x128 f16 swizzled
  unsigned short* wh16 = wi16 + 512 * 128;                     // 512x128
  unsigned short* gq16 = wh16 + 512 * 128;                     // 128x128
  unsigned short* pq16 = gq16 + 128 * 128;                     // 128x128

  // Output offsets: probs (S,B,S) | sels (S,B) | hx (B,H) | cx (B,H)
  const size_t OFF_SELS = (size_t)Ssz * Bsz * Ssz;
  const size_t OFF_HX   = OFF_SELS + (size_t)Ssz * Bsz;
  const size_t OFF_CX   = OFF_HX + N_BH;

  // Reset state every call (deterministic under graph replay).
  init_mask_kernel<<<(Bsz * Ssz + 255) / 256, 256, 0, stream>>>(mask);

  // One-time weight swizzle to f16 WMMA layout.
  swizzle_w_kernel<<<16, 256, 0, stream>>>(Wi,   wi16, 128);  // 32 n-tiles * 4
  swizzle_w_kernel<<<16, 256, 0, stream>>>(Wh,   wh16, 128);
  swizzle_w_kernel<<< 4, 256, 0, stream>>>(g_Wq, gq16, 32);   // 8 n-tiles * 4
  swizzle_w_kernel<<< 4, 256, 0, stream>>>(p_Wq, pq16, 32);

  // One-time precompute of e_g / e_p (resident in L2 afterwards).
  precompute_e_kernel<<<dim3(Ssz, Bsz / 32, 2), 256, 0, stream>>>(
      context, g_Wref, g_bref, p_Wref, p_bref, eg, ep);

  float* hbuf[2] = {hb0, hb1};
  float* cbuf[2] = {cb0, cb1};

  for (int t = 0; t < Ssz; ++t) {
    const float* xin = (t == 0) ? decoder_input : xbuf;
    const float* hin = (t == 0) ? h0 : hbuf[(t - 1) & 1];
    const float* cin = (t == 0) ? c0 : cbuf[(t - 1) & 1];
    float* hy = hbuf[t & 1];
    float* cy = cbuf[t & 1];

    lstm_kernel<<<64, 256, 0, stream>>>(xin, hin, cin, wi16, bi, wh16, bh, hy, cy);

    // glimpse head
    gemm128_bias_kernel<<<64, 256, 0, stream>>>(hy, gq16, g_bq, qbuf);
    glimpse_kernel<<<Bsz, 128, 0, stream>>>(qbuf, eg, g_v, mask, gl);

    // pointer head
    gemm128_bias_kernel<<<64, 256, 0, stream>>>(gl, pq16, p_bq, qbuf);
    pointer_kernel<<<Bsz, 128, 0, stream>>>(qbuf, ep, p_v, mask, embedded,
                                            xbuf, out, out + OFF_SELS, t);
  }

  // Final hidden / cell state (last write was t=99 -> buffers [1]).
  hipMemcpyAsync(out + OFF_HX, hbuf[1], N_BH * sizeof(float),
                 hipMemcpyDeviceToDevice, stream);
  hipMemcpyAsync(out + OFF_CX, cbuf[1], N_BH * sizeof(float),
                 hipMemcpyDeviceToDevice, stream);
}
